// MHA_76072460747305
// MI455X (gfx1250) — matrix operand order repaired and checked by reading, not compiled, not run on hardware
//
#include <hip/hip_runtime.h>
#include <cstdint>

#define D_MODEL 1024
#define N_HEADS 16
#define D_K     64
#define BATCH   2
#define SEQ     2048
#define MROWS   (BATCH * SEQ)   // 4096

typedef __attribute__((ext_vector_type(16))) __bf16 v16bf;
typedef __attribute__((ext_vector_type(8)))  float  v8f;
typedef __attribute__((ext_vector_type(4)))  unsigned int v4u;
typedef __attribute__((ext_vector_type(8)))  int v8i;
typedef __attribute__((ext_vector_type(4)))  int v4i;

union Frag16 {
    uint4 u[2];   // 32 bytes
    v16bf v;      // 16 x bf16
};

__device__ __forceinline__ unsigned short f2bf(float f) {
    union { float f; unsigned int u; } x;
    x.f = f;
    unsigned int u = x.u;
    u += 0x7FFFu + ((u >> 16) & 1u);   // round-to-nearest-even
    return (unsigned short)(u >> 16);
}

// ---- TDM: stage a [rows=64] x [cols=128] bf16 tile (row stride = k_stride elems)
// from global into LDS at byte offset lds_off. 2D descriptor per CDNA5 D# layout.
__device__ __forceinline__ void tdm_load_tile_64x128(const unsigned short* gptr,
                                                     unsigned lds_off, int k_stride) {
    unsigned long long ga = (unsigned long long)(size_t)gptr;
    v4u g0;
    g0[0] = 1u;                                   // count=1, user descriptor
    g0[1] = lds_off;                              // lds_addr
    g0[2] = (unsigned)(ga & 0xFFFFFFFFu);         // global_addr[31:0]
    g0[3] = (unsigned)((ga >> 32) & 0x01FFFFFFu)  // global_addr[56:32]
            | 0x80000000u;                        // type=2 ("image")
    const unsigned dim0 = (unsigned)k_stride;     // tensor_dim0 (k, elems)
    const unsigned dim1 = 1024u;                  // tensor_dim1 (rows)
    v8i g1;
    g1[0] = 0x00010000;                           // workgroup_mask=0, data_size=1 (2B)
    g1[1] = (int)((dim0 & 0xFFFFu) << 16);        // [31:16]=tensor_dim0 lo
    g1[2] = (int)(((dim0 >> 16) & 0xFFFFu) | ((dim1 & 0xFFFFu) << 16));
    g1[3] = (int)(((dim1 >> 16) & 0xFFFFu) | (128u << 16));  // tile_dim0 = 128
    g1[4] = 64;                                   // tile_dim1 = 64, tile_dim2 = 0
    g1[5] = k_stride;                             // tensor_dim0_stride[31:0]
    g1[6] = 0;                                    // stride0 hi | stride1 lo
    g1[7] = 0;                                    // stride1 hi
    v4i z4 = {0, 0, 0, 0};
#if defined(__clang_major__) && (__clang_major__ >= 23)
    v8i z8 = {0, 0, 0, 0, 0, 0, 0, 0};
    __builtin_amdgcn_tensor_load_to_lds(g0, g1, z4, z4, z8, 0);
#else
    __builtin_amdgcn_tensor_load_to_lds(g0, g1, z4, z4, 0);
#endif
}

// ---------------- f32 -> bf16 conversion (x4 vectorized) ----------------
__global__ __launch_bounds__(256)
void cvt_bf16_kernel(const float* __restrict__ src, unsigned short* __restrict__ dst, int n4) {
    int i = blockIdx.x * blockDim.x + threadIdx.x;
    if (i >= n4) return;
    float4 f = reinterpret_cast<const float4*>(src)[i];
    ushort4 o;
    o.x = f2bf(f.x); o.y = f2bf(f.y); o.z = f2bf(f.z); o.w = f2bf(f.w);
    reinterpret_cast<ushort4*>(dst)[i] = o;
}

// ---------------- GEMM: C[M,N] = A[M,K] @ W[N,K]^T + bias[N] ----------------
// Block = 8 waves -> 128x64 C tile. W tiles staged into LDS by the Tensor Data
// Mover (double-buffered); A fragments double-buffered from global; B-frag LDS
// loads interleaved with WMMAs for partial-counter waits.
// mode 0: bf16 head-split [B,H,T,D_K] | mode 1: bf16 transposed [B,H,D_K,T]
// mode 2: f32 row-major [M,N]
__global__ __launch_bounds__(256)
void gemm_xwt_kernel(const unsigned short* __restrict__ A,
                     const unsigned short* __restrict__ W,
                     const float* __restrict__ bias,
                     unsigned short* __restrict__ outb,
                     float* __restrict__ outf,
                     int M, int N, int K, int mode) {
    __shared__ unsigned short Wt[2][64][128];   // 32 KB double buffer

    const int lane = threadIdx.x & 31;
    const int wave = threadIdx.x >> 5;
    const int nl = lane & 15;
    const int g  = lane >> 4;

    const int nblk = N / 64;
    const int bm = blockIdx.x / nblk;
    const int bn = blockIdx.x % nblk;
    const int m0 = bm * 128 + wave * 16;
    const int n0 = bn * 64;

    v8f acc[4];
#pragma unroll
    for (int f = 0; f < 4; ++f)
#pragma unroll
        for (int r = 0; r < 8; ++r) acc[f][r] = 0.f;

    const unsigned short* Arow = A + (size_t)(m0 + nl) * K;
    const unsigned short* Wtile0 = W + (size_t)n0 * K;

    Frag16 afA[4], afB[4];
    auto loadA = [&](int stage, Frag16 (&dst)[4]) {
        const int ks = stage * 128;
#pragma unroll
        for (int kk = 0; kk < 4; ++kk) {
            dst[kk].u[0] = *reinterpret_cast<const uint4*>(Arow + ks + kk * 32 + g * 16);
            dst[kk].u[1] = *reinterpret_cast<const uint4*>(Arow + ks + kk * 32 + g * 16 + 8);
        }
    };
    // compute one 128-wide k stage: interleave B-frag LDS loads with WMMAs
    auto compute = [&](Frag16 (&af4)[4], int lbuf) {
#pragma unroll
        for (int kk = 0; kk < 4; ++kk) {
            const unsigned short* base = &Wt[lbuf][nl][kk * 32 + g * 16];
            Frag16 b0, b1, b2, b3;
            b0.u[0] = *reinterpret_cast<const uint4*>(base);
            b0.u[1] = *reinterpret_cast<const uint4*>(base + 8);
            b1.u[0] = *reinterpret_cast<const uint4*>(base + 16 * 128);
            b1.u[1] = *reinterpret_cast<const uint4*>(base + 16 * 128 + 8);
            acc[0] = __builtin_amdgcn_wmma_f32_16x16x32_bf16(
                false, af4[kk].v, false, b0.v, (short)0, acc[0], false, false);
            b2.u[0] = *reinterpret_cast<const uint4*>(base + 32 * 128);
            b2.u[1] = *reinterpret_cast<const uint4*>(base + 32 * 128 + 8);
            acc[1] = __builtin_amdgcn_wmma_f32_16x16x32_bf16(
                false, af4[kk].v, false, b1.v, (short)0, acc[1], false, false);
            b3.u[0] = *reinterpret_cast<const uint4*>(base + 48 * 128);
            b3.u[1] = *reinterpret_cast<const uint4*>(base + 48 * 128 + 8);
            acc[2] = __builtin_amdgcn_wmma_f32_16x16x32_bf16(
                false, af4[kk].v, false, b2.v, (short)0, acc[2], false, false);
            acc[3] = __builtin_amdgcn_wmma_f32_16x16x32_bf16(
                false, af4[kk].v, false, b3.v, (short)0, acc[3], false, false);
        }
    };

    // prologue: stage 0 -> LDS buffer 0, A fragments for stage 0
    loadA(0, afA);
    if (wave == 0) {
        tdm_load_tile_64x128(Wtile0, (unsigned)(size_t)&Wt[0][0][0], K);
        __builtin_amdgcn_s_wait_tensorcnt(0);
    }
    __syncthreads();

    const int nstages = K / 128;   // even (K = 1024 -> 8)
    for (int stage = 0; stage < nstages; stage += 2) {
        // even stage: compute(afA, buf0); preload afB / buf1
        if (stage + 1 < nstages) {
            if (wave == 0)
                tdm_load_tile_64x128(Wtile0 + (stage + 1) * 128,
                                     (unsigned)(size_t)&Wt[1][0][0], K);
            loadA(stage + 1, afB);
        }
        compute(afA, 0);
        if (wave == 0 && stage + 1 < nstages)
            __builtin_amdgcn_s_wait_tensorcnt(0);
        __syncthreads();
        if (stage + 1 >= nstages) break;

        // odd stage: compute(afB, buf1); preload afA / buf0
        if (stage + 2 < nstages) {
            if (wave == 0)
                tdm_load_tile_64x128(Wtile0 + (stage + 2) * 128,
                                     (unsigned)(size_t)&Wt[0][0][0], K);
            loadA(stage + 2, afA);
        }
        compute(afB, 1);
        if (wave == 0 && stage + 2 < nstages)
            __builtin_amdgcn_s_wait_tensorcnt(0);
        __syncthreads();
    }

#pragma unroll
    for (int f = 0; f < 4; ++f) {
        const int n = n0 + f * 16 + nl;
        const float bv = bias ? bias[n] : 0.f;
#pragma unroll
        for (int r = 0; r < 8; ++r) {
            const int m = m0 + r + g * 8;        // C layout: row = r + g*8
            const float val = acc[f][r] + bv;
            if (mode == 2) {
                outf[(size_t)m * N + n] = val;
            } else {
                const int bb = m / SEQ, t = m % SEQ;
                const int h = n / D_K, d = n % D_K;
                if (mode == 0)
                    outb[(((size_t)(bb * N_HEADS + h)) * SEQ + t) * D_K + d] = f2bf(val);
                else
                    outb[(((size_t)(bb * N_HEADS + h)) * D_K + d) * SEQ + t] = f2bf(val);
            }
        }
    }
}

// ---------------- Flash attention: one wave per 16-row query tile ----------------
__global__ __launch_bounds__(256)
void attn_kernel(const unsigned short* __restrict__ Qh,
                 const unsigned short* __restrict__ Kh,
                 const unsigned short* __restrict__ Vt,
                 const int* __restrict__ maskp,
                 const int* __restrict__ causalp,
                 unsigned short* __restrict__ O) {
    __shared__ unsigned short Pst[8][16][32];   // 1 KB per wave

    const int lane = threadIdx.x & 31;
    const int wave = threadIdx.x >> 5;
    const int nl = lane & 15;
    const int g  = lane >> 4;

    const int qtiles = SEQ / 16;                 // 128
    const int tile = blockIdx.x * 8 + wave;      // 0..4095
    const int qt = tile % qtiles;
    const int bh = tile / qtiles;                // 0..31
    const int b = bh / N_HEADS, h = bh % N_HEADS;
    const int q0 = qt * 16;
    const int causal = causalp[0];

    const unsigned short* Qb = Qh + (size_t)bh * SEQ * D_K;
    const unsigned short* Kb = Kh + (size_t)bh * SEQ * D_K;
    const unsigned short* Vb = Vt + (size_t)bh * D_K * SEQ;

    // Q fragments for the whole tile (d = 0..31, 32..63), reused every k block
    Frag16 qa[2];
    {
        const unsigned short* qrow = Qb + (size_t)(q0 + nl) * D_K;
#pragma unroll
        for (int j = 0; j < 2; ++j) {
            qa[j].u[0] = *reinterpret_cast<const uint4*>(qrow + j * 32 + g * 16);
            qa[j].u[1] = *reinterpret_cast<const uint4*>(qrow + j * 32 + g * 16 + 8);
        }
    }

    v8f acc[4];
    float mrow[8], lrow[8];
#pragma unroll
    for (int r = 0; r < 8; ++r) {
        mrow[r] = -__builtin_inff();
        lrow[r] = 0.f;
#pragma unroll
        for (int f = 0; f < 4; ++f) acc[f][r] = 0.f;
    }

    const int hi = causal ? (q0 + 16) : SEQ;
    const float scale = 0.125f;                  // 1/sqrt(64)

    for (int k0 = 0; k0 < hi; k0 += 32) {
        // prefetch next key block into near cache while we compute this one
        if (k0 + 32 < hi) {
            __builtin_prefetch(Kb + (size_t)(k0 + 32 + nl) * D_K, 0, 3);
            __builtin_prefetch(Kb + (size_t)(k0 + 48 + nl) * D_K, 0, 3);
        }

        // ---- issue ALL global loads for this block up front:
        // 4 K fragments, 4 V fragments (V doesn't depend on P), 2 mask words
        Frag16 kf[2][2];
#pragma unroll
        for (int sub = 0; sub < 2; ++sub) {
            const unsigned short* krow = Kb + (size_t)(k0 + sub * 16 + nl) * D_K;
#pragma unroll
            for (int j = 0; j < 2; ++j) {
                kf[sub][j].u[0] = *reinterpret_cast<const uint4*>(krow + j * 32 + g * 16);
                kf[sub][j].u[1] = *reinterpret_cast<const uint4*>(krow + j * 32 + g * 16 + 8);
            }
        }
        Frag16 vf[4];
#pragma unroll
        for (int f = 0; f < 4; ++f) {
            const unsigned short* vrow = Vb + (size_t)(f * 16 + nl) * SEQ + k0 + g * 16;
            vf[f].u[0] = *reinterpret_cast<const uint4*>(vrow);
            vf[f].u[1] = *reinterpret_cast<const uint4*>(vrow + 8);
        }
        int msk[2];
        msk[0] = maskp[b * SEQ + k0 + nl];
        msk[1] = maskp[b * SEQ + k0 + 16 + nl];

        // ---- scores: 4 WMMAs ----
        v8f s[2];
#pragma unroll
        for (int sub = 0; sub < 2; ++sub) {
#pragma unroll
            for (int r = 0; r < 8; ++r) s[sub][r] = 0.f;
#pragma unroll
            for (int j = 0; j < 2; ++j)
                s[sub] = __builtin_amdgcn_wmma_f32_16x16x32_bf16(
                    false, qa[j].v, false, kf[sub][j].v, (short)0, s[sub], false, false);
        }

        // ---- scale + key mask + causal mask ----
#pragma unroll
        for (int sub = 0; sub < 2; ++sub) {
            const int col = k0 + sub * 16 + nl;
#pragma unroll
            for (int r = 0; r < 8; ++r) {
                const int row = q0 + r + g * 8;
                float v = s[sub][r] * scale;
                if (msk[sub] == 0 || (causal && col > row)) v = -__builtin_inff();
                s[sub][r] = v;
            }
        }

        // ---- online softmax update (V loads in flight underneath) ----
#pragma unroll
        for (int r = 0; r < 8; ++r) {
            float rm = fmaxf(s[0][r], s[1][r]);
#pragma unroll
            for (int off = 8; off >= 1; off >>= 1)
                rm = fmaxf(rm, __shfl_xor(rm, off, 32));
            const float mnew = fmaxf(mrow[r], rm);
            float sc, p0, p1;
            if (mnew == -__builtin_inff()) {
                sc = 1.f; p0 = 0.f; p1 = 0.f;    // fully-masked so far
            } else {
                sc = __expf(mrow[r] - mnew);     // -inf -> 0
                p0 = __expf(s[0][r] - mnew);
                p1 = __expf(s[1][r] - mnew);
            }
            float ps = p0 + p1;
#pragma unroll
            for (int off = 8; off >= 1; off >>= 1)
                ps += __shfl_xor(ps, off, 32);
            lrow[r] = lrow[r] * sc + ps;
            mrow[r] = mnew;
#pragma unroll
            for (int f = 0; f < 4; ++f) acc[f][r] *= sc;
            // stash P (C layout -> LDS row-major 16x32)
            Pst[wave][r + g * 8][nl]      = f2bf(p0);
            Pst[wave][r + g * 8][16 + nl] = f2bf(p1);
        }

        // DS ops are in-order per wave; fence the LDS handoff explicitly.
        asm volatile("s_wait_dscnt 0x0" ::: "memory");

        // ---- reload P as an A fragment; P @ V (V already resident) ----
        Frag16 pa;
        pa.u[0] = *reinterpret_cast<const uint4*>(&Pst[wave][nl][g * 16]);
        pa.u[1] = *reinterpret_cast<const uint4*>(&Pst[wave][nl][g * 16 + 8]);
#pragma unroll
        for (int f = 0; f < 4; ++f)
            acc[f] = __builtin_amdgcn_wmma_f32_16x16x32_bf16(
                false, pa.v, false, vf[f].v, (short)0, acc[f], false, false);
    }

    // ---- normalize and store concatenated-head output (bf16) ----
#pragma unroll
    for (int f = 0; f < 4; ++f) {
        const int d = f * 16 + nl;
#pragma unroll
        for (int r = 0; r < 8; ++r) {
            const int t = q0 + r + g * 8;
            const float l = lrow[r];
            const float val = (l > 0.f) ? acc[f][r] / l : 0.f;
            O[((size_t)(b * SEQ + t)) * D_MODEL + h * D_K + d] = f2bf(val);
        }
    }
}

// ---------------- host-side launch ----------------
extern "C" void kernel_launch(void* const* d_in, const int* in_sizes, int n_in,
                              void* d_out, int out_size, void* d_ws, size_t ws_size,
                              hipStream_t stream) {
    (void)in_sizes; (void)n_in; (void)out_size; (void)ws_size;

    const float* q  = (const float*)d_in[0];
    const float* k  = (const float*)d_in[1];
    const float* v  = (const float*)d_in[2];
    const float* Wq = (const float*)d_in[3];
    const float* bq = (const float*)d_in[4];
    const float* Wk = (const float*)d_in[5];
    const float* bk = (const float*)d_in[6];
    const float* Wv = (const float*)d_in[7];
    const float* bv = (const float*)d_in[8];
    const float* Wo = (const float*)d_in[9];
    const float* bo = (const float*)d_in[10];
    const int* mask   = (const int*)d_in[11];
    const int* causal = (const int*)d_in[12];
    float* out = (float*)d_out;

    const size_t SZ_X = (size_t)MROWS * D_MODEL;     // 4M elems
    const size_t SZ_W = (size_t)D_MODEL * D_MODEL;   // 1M elems
    unsigned short* qb  = (unsigned short*)d_ws;
    unsigned short* kb  = qb  + SZ_X;
    unsigned short* vb  = kb  + SZ_X;
    unsigned short* Wqb = vb  + SZ_X;
    unsigned short* Wkb = Wqb + SZ_W;
    unsigned short* Wvb = Wkb + SZ_W;
    unsigned short* Wob = Wvb + SZ_W;
    unsigned short* Qh  = Wob + SZ_W;
    unsigned short* Kh  = Qh  + SZ_X;
    unsigned short* Vt  = Kh  + SZ_X;
    unsigned short* Ob  = Vt  + SZ_X;                // total 64 MiB of d_ws

    dim3 blk(256);
    const int gX = (int)((SZ_X / 4 + 255) / 256);    // 4096 blocks
    const int gW = (int)((SZ_W / 4 + 255) / 256);    // 1024 blocks
    cvt_bf16_kernel<<<gX, blk, 0, stream>>>(q,  qb,  (int)(SZ_X / 4));
    cvt_bf16_kernel<<<gX, blk, 0, stream>>>(k,  kb,  (int)(SZ_X / 4));
    cvt_bf16_kernel<<<gX, blk, 0, stream>>>(v,  vb,  (int)(SZ_X / 4));
    cvt_bf16_kernel<<<gW, blk, 0, stream>>>(Wq, Wqb, (int)(SZ_W / 4));
    cvt_bf16_kernel<<<gW, blk, 0, stream>>>(Wk, Wkb, (int)(SZ_W / 4));
    cvt_bf16_kernel<<<gW, blk, 0, stream>>>(Wv, Wvb, (int)(SZ_W / 4));
    cvt_bf16_kernel<<<gW, blk, 0, stream>>>(Wo, Wob, (int)(SZ_W / 4));

    const int gemmBlocks = (MROWS / 128) * (D_MODEL / 64);        // 512
    gemm_xwt_kernel<<<gemmBlocks, blk, 0, stream>>>(qb, Wqb, bq, Qh, nullptr,
                                                    MROWS, D_MODEL, D_MODEL, 0);
    gemm_xwt_kernel<<<gemmBlocks, blk, 0, stream>>>(kb, Wkb, bk, Kh, nullptr,
                                                    MROWS, D_MODEL, D_MODEL, 0);
    gemm_xwt_kernel<<<gemmBlocks, blk, 0, stream>>>(vb, Wvb, bv, Vt, nullptr,
                                                    MROWS, D_MODEL, D_MODEL, 1);

    const int attnBlocks = (BATCH * N_HEADS * (SEQ / 16)) / 8;    // 512
    attn_kernel<<<attnBlocks, blk, 0, stream>>>(Qh, Kh, Vt, mask, causal, Ob);

    gemm_xwt_kernel<<<gemmBlocks, blk, 0, stream>>>(Ob, Wob, bo, nullptr, out,
                                                    MROWS, D_MODEL, D_MODEL, 2);
}
